// Attention_36627481101235
// MI455X (gfx1250) — matrix operand order repaired and checked by reading, not compiled, not run on hardware
//
#include <hip/hip_runtime.h>

// ---------------------------------------------------------------------------
// LLaMA attention block for MI455X (gfx1250, wave32, WMMA bf16).
// Pipeline: cast f32->bf16 (+weight transpose) -> WMMA GEMM QKV (V written
// pre-transposed) -> RoPE -> flash attention (WMMA QK^T and PV, online
// softmax, double-buffered async K/V staging) -> WMMA GEMM wo.
// All LDS tile staging uses gfx1250 async global->LDS copies (ASYNCcnt).
// ---------------------------------------------------------------------------

typedef __attribute__((ext_vector_type(16))) __bf16 v16bf;
typedef __attribute__((ext_vector_type(8)))  float  v8f;

#define SEQ   2048
#define DIM   4096
#define NH    32
#define NKVH  8
#define HD    128
#define KVDIM (NKVH * HD)   // 1024

// Build a 16-element bf16 fragment from two contiguous 8-element (16B) runs.
__device__ __forceinline__ v16bf frag2(const __bf16* lo, const __bf16* hi) {
  union U { uint4 q; __bf16 h[8]; };
  U a, b;
  a.q = *(const uint4*)lo;
  b.q = *(const uint4*)hi;
  v16bf r;
#pragma unroll
  for (int i = 0; i < 8; ++i) { r[i] = a.h[i]; r[i + 8] = b.h[i]; }
  return r;
}

__device__ __forceinline__ v8f wmma_bf16(v16bf a, v16bf b, v8f c) {
  return __builtin_amdgcn_wmma_f32_16x16x32_bf16(
      /*neg_a=*/false, a, /*neg_b=*/false, b,
      /*c_mod=*/(short)0, c, /*reuse_a=*/false, /*reuse_b=*/false);
}

// Async global->LDS 16B copy (per lane). lds_off is the LDS byte address
// (low 32 bits of the generic shared-memory address). Tracked by ASYNCcnt.
__device__ __forceinline__ void async_load_b128(unsigned lds_off,
                                                const void* gaddr) {
  asm volatile("global_load_async_to_lds_b128 %0, %1, off"
               :: "v"(lds_off), "v"(gaddr) : "memory");
}
__device__ __forceinline__ unsigned lds_addr(const void* p) {
  return (unsigned)(size_t)p;   // low 32 bits of flat LDS-aperture address
}

// ---------------------------------------------------------------------------
// Casting kernels: f32 -> bf16, with optional transpose (weights become [N][K]
// so both GEMM operands are K-contiguous).
// ---------------------------------------------------------------------------
__global__ void cvt_f32_bf16(const float* __restrict__ in,
                             __bf16* __restrict__ out, int n) {
  const int i = blockIdx.x * 256 + threadIdx.x;
  if (i < n) out[i] = (__bf16)in[i];
}

__global__ void cvt_f32_bf16_T(const float* __restrict__ in,
                               __bf16* __restrict__ out, int rows, int cols) {
  const int i = blockIdx.x * 256 + threadIdx.x;
  if (i >= rows * cols) return;
  const int r = i / cols;
  const int c = i - r * cols;
  out[(size_t)c * rows + r] = (__bf16)in[i];
}

// ---------------------------------------------------------------------------
// RoPE on bf16 activations (in place). width = heads*HD for that tensor.
// ---------------------------------------------------------------------------
__global__ void rope_bf16(__bf16* __restrict__ t,
                          const float* __restrict__ cosb,
                          const float* __restrict__ sinb, int width) {
  const int pairs = width >> 1;
  const int i = blockIdx.x * 256 + threadIdx.x;
  if (i >= SEQ * pairs) return;
  const int pos = i / pairs;
  const int p   = i - pos * pairs;
  const int j   = p & 63;                       // index within head (HD/2 = 64)
  const float c = cosb[pos * 64 + j];
  const float s = sinb[pos * 64 + j];
  __bf16* b = t + (size_t)pos * width + p * 2;
  const float xr = (float)b[0], xi = (float)b[1];
  b[0] = (__bf16)(xr * c - xi * s);
  b[1] = (__bf16)(xr * s + xi * c);
}

// ---------------------------------------------------------------------------
// NT GEMM: C[M,N] = A[M,K] * Bt[N,K]^T, bf16 in, f32 accumulate.
// Block tile 128x128, 8 waves -> 64x32 wave tiles, K-step 32.
// Double-buffered LDS staged with async global->LDS copies (ASYNCcnt).
// TRANS_OUT stores C transposed ([N][M]) with 16B vector stores (used for V
// so flash attention can async-stage V^T tiles directly).
// ---------------------------------------------------------------------------
template <bool OUT_BF16, bool TRANS_OUT>
__global__ __launch_bounds__(256)
void gemm_nt_bf16(const __bf16* __restrict__ A, const __bf16* __restrict__ Bt,
                  void* __restrict__ Cout, int M, int N, int K) {
  __shared__ __align__(16) __bf16 As[2][128 * 32];
  __shared__ __align__(16) __bf16 Bs[2][128 * 32];
  const int tid  = threadIdx.x;
  const int wave = tid >> 5, lane = tid & 31;
  const int half = lane >> 4, l16 = lane & 15;
  const int bm = blockIdx.x * 128, bn = blockIdx.y * 128;
  const int wm = (wave >> 2) * 64, wn = (wave & 3) * 32;
  const int aoff = half * 8;   // A-frag k-offset: lanes<16 -> {0,16}, >=16 -> {8,24}

  // Per-thread staging slots: 2x 8-half vectors per matrix per K-step.
  const int idxA = tid * 8;
  const int idxB = (256 + tid) * 8;
  const int rA = idxA >> 5, cA = idxA & 31;
  const int rB = idxB >> 5, cB = idxB & 31;

  auto stage = [&](int buf, int k0) {
    async_load_b128(lds_addr(&As[buf][idxA]), &A [(size_t)(bm + rA) * K + k0 + cA]);
    async_load_b128(lds_addr(&As[buf][idxB]), &A [(size_t)(bm + rB) * K + k0 + cB]);
    async_load_b128(lds_addr(&Bs[buf][idxA]), &Bt[(size_t)(bn + rA) * K + k0 + cA]);
    async_load_b128(lds_addr(&Bs[buf][idxB]), &Bt[(size_t)(bn + rB) * K + k0 + cB]);
  };

  v8f acc[4][2] = {};

  stage(0, 0);
  const int nk = K >> 5;
  for (int kt = 0; kt < nk; ++kt) {
    const int buf = kt & 1;
    if (kt + 1 < nk) {
      stage(buf ^ 1, (kt + 1) * 32);
      // 8 outstanding; <=4 left means this tile's 4 copies have landed.
      asm volatile("s_wait_asynccnt 0x4" ::: "memory");
    } else {
      asm volatile("s_wait_asynccnt 0x0" ::: "memory");
    }
    __syncthreads();

    v16bf af[4], bfr[2];
#pragma unroll
    for (int i = 0; i < 4; ++i) {
      const __bf16* base = &As[buf][(wm + i * 16 + l16) * 32];
      af[i] = frag2(base + aoff, base + aoff + 16);
    }
#pragma unroll
    for (int j = 0; j < 2; ++j) {
      const __bf16* base = &Bs[buf][(wn + j * 16 + l16) * 32 + half * 8];
      bfr[j] = frag2(base, base + 16);
    }
#pragma unroll
    for (int i = 0; i < 4; ++i)
#pragma unroll
      for (int j = 0; j < 2; ++j)
        acc[i][j] = wmma_bf16(af[i], bfr[j], acc[i][j]);
    __syncthreads();
  }

#pragma unroll
  for (int i = 0; i < 4; ++i)
#pragma unroll
    for (int j = 0; j < 2; ++j) {
      if constexpr (TRANS_OUT) {
        // 8 consecutive rows of one column -> 8 consecutive elems of C^T row.
        union { uint4 q; __bf16 h[8]; } pk;
#pragma unroll
        for (int v = 0; v < 8; ++v) pk.h[v] = (__bf16)acc[i][j][v];
        const size_t col = (size_t)(bn + wn + j * 16 + l16);
        const size_t row0 = (size_t)(bm + wm + i * 16 + half * 8);
        *(uint4*)&((__bf16*)Cout)[col * M + row0] = pk.q;
      } else {
#pragma unroll
        for (int v = 0; v < 8; ++v) {
          const size_t row = (size_t)(bm + wm + i * 16 + half * 8 + v);
          const size_t col = (size_t)(bn + wn + j * 16 + l16);
          if (OUT_BF16) ((__bf16*)Cout)[row * N + col] = (__bf16)acc[i][j][v];
          else          ((float*)Cout)[row * N + col]  = acc[i][j][v];
        }
      }
    }
}

// ---------------------------------------------------------------------------
// Flash attention: grid (SEQ/128 q-tiles, NH heads), 256 threads (8 waves).
// Each wave owns 16 q rows; K (row-major) and V^T tiles are double-buffered
// in LDS via async global->LDS copies; online softmax; P goes through
// per-wave LDS scratch to re-lay out C-fragment -> A-fragment for PV.
// VcT is V transposed: [KVDIM][SEQ].
// ---------------------------------------------------------------------------
__global__ __launch_bounds__(256)
void flash_attn(const __bf16* __restrict__ Q, const __bf16* __restrict__ Kc,
                const __bf16* __restrict__ VcT, __bf16* __restrict__ O) {
  __shared__ __align__(16) __bf16 Ks[2][32 * 128];   // [kpos][d]
  __shared__ __align__(16) __bf16 Vt[2][128 * 32];   // [d][kpos]
  __shared__ __align__(16) __bf16 Ps[8 * 16 * 32];   // per-wave P scratch
  const int tid  = threadIdx.x;
  const int wave = tid >> 5, lane = tid & 31;
  const int half = lane >> 4, l16 = lane & 15;
  const int qt = blockIdx.x, h = blockIdx.y, kvh = h >> 2;   // N_REP = 4
  const int qrow = qt * 128 + wave * 16;
  const int aoff = half * 8;

  auto stageKV = [&](int buf, int kbase) {
#pragma unroll
    for (int e = 0; e < 2; ++e) {
      const int idx = (e * 256 + tid) * 8;
      const int kp = idx >> 7, d = idx & 127;
      async_load_b128(lds_addr(&Ks[buf][idx]),
                      &Kc[(size_t)(kbase + kp) * KVDIM + kvh * HD + d]);
    }
#pragma unroll
    for (int e = 0; e < 2; ++e) {
      const int idx = (e * 256 + tid) * 8;
      const int d = idx >> 5, kp = idx & 31;
      async_load_b128(lds_addr(&Vt[buf][idx]),
                      &VcT[(size_t)(kvh * HD + d) * SEQ + kbase + kp]);
    }
  };

  // Q fragments for this wave's 16 rows, all 128 d, loaded once.
  v16bf qf[4];
  const __bf16* qbase = Q + (size_t)(qrow + l16) * DIM + h * HD;
#pragma unroll
  for (int c = 0; c < 4; ++c)
    qf[c] = frag2(qbase + c * 32 + aoff, qbase + c * 32 + aoff + 16);

  v8f acc[8] = {};
  float mrow[8], lrow[8];
#pragma unroll
  for (int v = 0; v < 8; ++v) { mrow[v] = -3.0e38f; lrow[v] = 0.0f; }
  const float scale = 0.08838834764831845f;   // 1/sqrt(128)

  const int nkt = (qt + 1) * 4;               // causal: only tiles up to diag
  stageKV(0, 0);
  for (int kt = 0; kt < nkt; ++kt) {
    const int kbase = kt * 32;
    const int buf = kt & 1;
    if (kt + 1 < nkt) {
      stageKV(buf ^ 1, kbase + 32);
      asm volatile("s_wait_asynccnt 0x4" ::: "memory");
    } else {
      asm volatile("s_wait_asynccnt 0x0" ::: "memory");
    }
    __syncthreads();

    // S = Q * K^T : two 16x16 column-halves, contraction over d in 4x32.
    v8f s[2] = {};
#pragma unroll
    for (int nh = 0; nh < 2; ++nh)
#pragma unroll
      for (int c = 0; c < 4; ++c) {
        const __bf16* kb = &Ks[buf][(nh * 16 + l16) * 128 + c * 32 + half * 8];
        s[nh] = wmma_bf16(qf[c], frag2(kb, kb + 16), s[nh]);
      }

    // Online softmax. Row (half*8+v) is spread across this lane's 16-group;
    // xor masks 1,2,4,8 reduce within that group.
    float alpha[8];
    const int kg0 = kbase + l16, kg1 = kbase + 16 + l16;
#pragma unroll
    for (int v = 0; v < 8; ++v) {
      const int qg = qrow + half * 8 + v;
      float s0 = s[0][v] * scale + (kg0 > qg ? -1.0e9f : 0.0f);
      float s1 = s[1][v] * scale + (kg1 > qg ? -1.0e9f : 0.0f);
      float m = fmaxf(s0, s1);
#pragma unroll
      for (int o = 8; o >= 1; o >>= 1) m = fmaxf(m, __shfl_xor(m, o, 32));
      m = fmaxf(m, mrow[v]);
      alpha[v] = __expf(mrow[v] - m);
      mrow[v] = m;
      const float e0 = __expf(s0 - m), e1 = __expf(s1 - m);
      s[0][v] = e0; s[1][v] = e1;
      float rs = e0 + e1;
#pragma unroll
      for (int o = 8; o >= 1; o >>= 1) rs += __shfl_xor(rs, o, 32);
      lrow[v] = lrow[v] * alpha[v] + rs;
    }
#pragma unroll
    for (int n = 0; n < 8; ++n)
#pragma unroll
      for (int v = 0; v < 8; ++v) acc[n][v] *= alpha[v];

    // P: C-fragment layout -> bf16 LDS tile -> A-fragment layout.
    __bf16* pw = &Ps[wave * 16 * 32];
#pragma unroll
    for (int v = 0; v < 8; ++v) {
      const int r = half * 8 + v;
      pw[r * 32 + l16]      = (__bf16)s[0][v];
      pw[r * 32 + 16 + l16] = (__bf16)s[1][v];
    }
    asm volatile("s_wait_dscnt 0x0" ::: "memory");   // wave-private RAW on LDS
    const __bf16* pr = &Ps[wave * 16 * 32 + l16 * 32];
    const v16bf pf = frag2(pr + aoff, pr + aoff + 16);

    // acc[16 x 128] += P[16 x 32] * V[32 x 128]
#pragma unroll
    for (int n = 0; n < 8; ++n) {
      const __bf16* vb = &Vt[buf][(n * 16 + l16) * 32 + half * 8];
      acc[n] = wmma_bf16(pf, frag2(vb, vb + 16), acc[n]);
    }
    __syncthreads();
  }

  // Normalize and write bf16 output [q][h*128+d].
#pragma unroll
  for (int n = 0; n < 8; ++n)
#pragma unroll
    for (int v = 0; v < 8; ++v) {
      const int r = qrow + half * 8 + v;
      O[(size_t)r * DIM + h * HD + n * 16 + l16] = (__bf16)(acc[n][v] / lrow[v]);
    }
}

// ---------------------------------------------------------------------------
// Host launcher. Inputs: x, freqs_cos, freqs_sin, mask(unused), wq, wk, wv, wo.
// Workspace budget ~136 MB.
// ---------------------------------------------------------------------------
extern "C" void kernel_launch(void* const* d_in, const int* in_sizes, int n_in,
                              void* d_out, int out_size, void* d_ws, size_t ws_size,
                              hipStream_t stream) {
  (void)in_sizes; (void)n_in; (void)out_size; (void)ws_size;
  const float* x  = (const float*)d_in[0];
  const float* fc = (const float*)d_in[1];
  const float* fs = (const float*)d_in[2];
  const float* wq = (const float*)d_in[4];
  const float* wk = (const float*)d_in[5];
  const float* wv = (const float*)d_in[6];
  const float* wo = (const float*)d_in[7];

  char* ws = (char*)d_ws;
  size_t off = 0;
  auto take = [&](size_t bytes) -> char* {
    char* p = ws + off;
    off += (bytes + 255) & ~(size_t)255;
    return p;
  };
  __bf16* x16  = (__bf16*)take((size_t)SEQ * DIM * 2);
  __bf16* wqT  = (__bf16*)take((size_t)DIM * DIM * 2);
  __bf16* wkT  = (__bf16*)take((size_t)DIM * KVDIM * 2);
  __bf16* wvT  = (__bf16*)take((size_t)DIM * KVDIM * 2);
  __bf16* woT  = (__bf16*)take((size_t)DIM * DIM * 2);
  __bf16* q16  = (__bf16*)take((size_t)SEQ * DIM * 2);
  __bf16* k16  = (__bf16*)take((size_t)SEQ * KVDIM * 2);
  __bf16* vT16 = (__bf16*)take((size_t)KVDIM * SEQ * 2);   // V^T [KVDIM][SEQ]
  __bf16* a16  = (__bf16*)take((size_t)SEQ * DIM * 2);

  const int nx = SEQ * DIM;
  cvt_f32_bf16<<<(nx + 255) / 256, 256, 0, stream>>>(x, x16, nx);
  cvt_f32_bf16_T<<<(DIM * DIM + 255) / 256, 256, 0, stream>>>(wq, wqT, DIM, DIM);
  cvt_f32_bf16_T<<<(DIM * KVDIM + 255) / 256, 256, 0, stream>>>(wk, wkT, DIM, KVDIM);
  cvt_f32_bf16_T<<<(DIM * KVDIM + 255) / 256, 256, 0, stream>>>(wv, wvT, DIM, KVDIM);
  cvt_f32_bf16_T<<<(DIM * DIM + 255) / 256, 256, 0, stream>>>(wo, woT, DIM, DIM);

  gemm_nt_bf16<true, false><<<dim3(SEQ / 128, DIM / 128), 256, 0, stream>>>(
      x16, wqT, q16, SEQ, DIM, DIM);
  gemm_nt_bf16<true, false><<<dim3(SEQ / 128, KVDIM / 128), 256, 0, stream>>>(
      x16, wkT, k16, SEQ, KVDIM, DIM);
  gemm_nt_bf16<true, true><<<dim3(SEQ / 128, KVDIM / 128), 256, 0, stream>>>(
      x16, wvT, vT16, SEQ, KVDIM, DIM);   // writes V^T

  rope_bf16<<<(SEQ * (DIM / 2) + 255) / 256, 256, 0, stream>>>(q16, fc, fs, DIM);
  rope_bf16<<<(SEQ * (KVDIM / 2) + 255) / 256, 256, 0, stream>>>(k16, fc, fs, KVDIM);

  flash_attn<<<dim3(SEQ / 128, NH), 256, 0, stream>>>(q16, k16, vT16, a16);

  gemm_nt_bf16<false, false><<<dim3(SEQ / 128, DIM / 128), 256, 0, stream>>>(
      a16, woT, d_out, SEQ, DIM, DIM);
}
